// InteractionHead_52312701666131
// MI455X (gfx1250) — compile-verified, matrix operand order checked by reading, not run on hardware
//
#include <hip/hip_runtime.h>
#include <hip/hip_bf16.h>
#include <stdint.h>

#define N_DET 256
#define T_GT  64
#define CF    576
#define REP   512
#define NC    600
#define NC_PAD 640
#define M_TOT (N_DET * N_DET)   // 65536 pair rows
#define BM 128
#define BN 64
#define BK 32
#define STR 80                  // LDS row stride (bytes): 16B-aligned, bank-conflict-free

typedef __attribute__((ext_vector_type(16))) __bf16 bf16x16;
typedef __attribute__((ext_vector_type(8)))  float  f32x8;

__device__ __forceinline__ unsigned short f2bf(float f) {
  unsigned u = __float_as_uint(f);
  u += 0x7FFFu + ((u >> 16) & 1u);      // round-to-nearest-even
  return (unsigned short)(u >> 16);
}

union Frag { uint4 u[2]; bf16x16 v; };

// ---------------------------------------------------------------- h_mask ----
__global__ void hmask_k(const float* __restrict__ scores, float* __restrict__ hm) {
  int n = threadIdx.x;                       // 256 threads, 1 block
  const float* s = scores + n * 80;
  float s0 = s[0];
  int ok = 1;
  for (int j = 1; j < 80; ++j) ok &= (s0 >= s[j]);   // argmax==0 (first-occurrence tie)
  hm[n] = ok ? 1.0f : 0.0f;
}

// ------------------------------------------------------------- IoU tables ---
__global__ void iou_k(const float* __restrict__ boxes, const float* __restrict__ bh,
                      const float* __restrict__ bo, float* __restrict__ ih,
                      float* __restrict__ io) {
  __shared__ float sh[T_GT * 4], so[T_GT * 4];
  int tid = threadIdx.x;                     // 256 threads, 1 block
  if (tid < T_GT * 4) { sh[tid] = bh[tid]; so[tid] = bo[tid]; }
  __syncthreads();
  float x1 = boxes[tid * 4 + 0], y1 = boxes[tid * 4 + 1];
  float x2 = boxes[tid * 4 + 2], y2 = boxes[tid * 4 + 3];
  float area = (x2 - x1) * (y2 - y1);
  for (int t = 0; t < T_GT; ++t) {
    {
      float bx1 = sh[t*4], by1 = sh[t*4+1], bx2 = sh[t*4+2], by2 = sh[t*4+3];
      float ab = (bx2 - bx1) * (by2 - by1);
      float w = fmaxf(fminf(x2, bx2) - fmaxf(x1, bx1), 0.0f);
      float h = fmaxf(fminf(y2, by2) - fmaxf(y1, by1), 0.0f);
      float inter = w * h;
      ih[tid * T_GT + t] = inter / (area + ab - inter + 1e-8f);
    }
    {
      float bx1 = so[t*4], by1 = so[t*4+1], bx2 = so[t*4+2], by2 = so[t*4+3];
      float ab = (bx2 - bx1) * (by2 - by1);
      float w = fmaxf(fminf(x2, bx2) - fmaxf(x1, bx1), 0.0f);
      float h = fmaxf(fminf(y2, by2) - fmaxf(y1, by1), 0.0f);
      float inter = w * h;
      io[tid * T_GT + t] = inter / (area + ab - inter + 1e-8f);
    }
  }
}

// -------------------------------------- weight transpose + f32->bf16 -------
// src: [K][N] f32 row-major; dst: [Npad][K] bf16 (rows >= N zero-filled)
__global__ void convT_k(const float* __restrict__ src, unsigned short* __restrict__ dst,
                        int K, int N, int Npad) {
  int idx = blockIdx.x * 256 + threadIdx.x;
  if (idx >= Npad * K) return;
  int n = idx / K, k = idx - n * K;
  float v = (n < N) ? src[(size_t)k * N + n] : 0.0f;
  dst[idx] = f2bf(v);
}

// -------------------------------------------------------- WMMA GEMM --------
// C[M x Nout] = act(A[M x K] @ W[K x Nout] + bias); A row-major, Wt = W^T bf16 [Nout][K].
// AF32: A is f32 in global (layer 1), else bf16. FINAL: f32 store * mask, Nout=600.
template <bool AF32, bool FINAL>
__global__ __launch_bounds__(256) void gemm_k(
    const float* __restrict__ Af, const unsigned short* __restrict__ Ab, const int K,
    const unsigned short* __restrict__ Wt, const float* __restrict__ bias,
    float* __restrict__ outF, unsigned short* __restrict__ outB,
    const float* __restrict__ hmask) {
  __shared__ __align__(16) unsigned char sA[BM * STR];
  __shared__ __align__(16) unsigned char sB[BN * STR];

  const int tid = threadIdx.x;
  const int l = tid & 31;
  const int w = tid >> 5;
  const int wm = w >> 1;                 // 0..3 -> 32-row slab
  const int wn = w & 1;                  // 0..1 -> 32-col slab
  const int m0 = blockIdx.y * BM;
  const int n0 = blockIdx.x * BN;
  const int lrow = l & 15;
  const int lhalf16 = (l >> 4) * 16;     // byte offset: lane-half K selection

  f32x8 acc[2][2];
#pragma unroll
  for (int i = 0; i < 2; ++i)
#pragma unroll
    for (int j = 0; j < 2; ++j) {
      f32x8 z = {0.f, 0.f, 0.f, 0.f, 0.f, 0.f, 0.f, 0.f};
      acc[i][j] = z;
    }

  for (int kk = 0; kk < K; kk += BK) {
    // stage A tile (BM x BK) as bf16
    if constexpr (AF32) {
#pragma unroll
      for (int i = 0; i < 4; ++i) {
        int id = i * 256 + tid;          // 1024 float4 chunks
        int row = id >> 3, cc = id & 7;
        const float4 f = *(const float4*)(Af + (size_t)(m0 + row) * K + kk + cc * 4);
        uint2 pk;
        pk.x = (unsigned)f2bf(f.x) | ((unsigned)f2bf(f.y) << 16);
        pk.y = (unsigned)f2bf(f.z) | ((unsigned)f2bf(f.w) << 16);
        *(uint2*)(sA + row * STR + cc * 8) = pk;
      }
    } else {
#pragma unroll
      for (int i = 0; i < 2; ++i) {
        int id = i * 256 + tid;          // 512 x 16B chunks
        int row = id >> 2, cc = id & 3;
        *(uint4*)(sA + row * STR + cc * 16) =
            *(const uint4*)(Ab + (size_t)(m0 + row) * K + kk + cc * 8);
      }
    }
    // stage B tile (BN x BK), Wt is N-major so K is contiguous
    {
      int row = tid >> 2, cc = tid & 3;  // 256 x 16B chunks
      *(uint4*)(sB + row * STR + cc * 16) =
          *(const uint4*)(Wt + (size_t)(n0 + row) * K + kk + cc * 8);
    }
    __syncthreads();

    Frag a[2], b[2];
#pragma unroll
    for (int mi = 0; mi < 2; ++mi) {     // ISA 16-bit A layout: two ds_load_b128
      const unsigned char* p = sA + (wm * 32 + mi * 16 + lrow) * STR + lhalf16;
      a[mi].u[0] = *(const uint4*)p;
      a[mi].u[1] = *(const uint4*)(p + 32);
    }
#pragma unroll
    for (int ni = 0; ni < 2; ++ni) {
      const unsigned char* p = sB + (wn * 32 + ni * 16 + lrow) * STR + lhalf16;
      b[ni].u[0] = *(const uint4*)p;
      b[ni].u[1] = *(const uint4*)(p + 32);
    }
#pragma unroll
    for (int mi = 0; mi < 2; ++mi)
#pragma unroll
      for (int ni = 0; ni < 2; ++ni)
        acc[mi][ni] = __builtin_amdgcn_wmma_f32_16x16x32_bf16(
            false, a[mi].v, false, b[ni].v, (short)0, acc[mi][ni], false, false);
    __syncthreads();
  }

  // epilogue (16x16 f32 C layout: lane n = l&15, VGPR r -> m = r + (l>>4)*8)
  const int lh8 = (l >> 4) * 8;
#pragma unroll
  for (int ni = 0; ni < 2; ++ni) {
    const int n = n0 + wn * 32 + ni * 16 + lrow;
    const float bv = (!FINAL || n < NC) ? bias[n] : 0.0f;
#pragma unroll
    for (int mi = 0; mi < 2; ++mi) {
#pragma unroll
      for (int r = 0; r < 8; ++r) {
        const int m = m0 + wm * 32 + mi * 16 + lh8 + r;
        float v = acc[mi][ni][r] + bv;
        if constexpr (FINAL) {
          if (n < NC) outF[(size_t)m * NC + n] = v * hmask[m >> 8];
        } else {
          v = v > 0.0f ? v : 0.0f;
          outB[(size_t)m * REP + n] = f2bf(v);
        }
      }
    }
  }
}

// ------------------------------------------------------------- labels ------
__global__ void labels_k(const float* __restrict__ ih, const float* __restrict__ io,
                         const int* __restrict__ hoi, const float* __restrict__ hm,
                         float* __restrict__ labels, float* __restrict__ maskO) {
  __shared__ float lab[NC];
  const int p = blockIdx.x, tid = threadIdx.x;   // 128 threads / pair
  const float m = hm[p >> 8];
  for (int j = tid; j < NC; j += 128) lab[j] = 0.0f;
  __syncthreads();
  if (m != 0.0f && tid < T_GT) {
    float v = fminf(ih[(p >> 8) * T_GT + tid], io[(p & 255) * T_GT + tid]);
    if (v >= 0.5f) lab[hoi[tid]] = 1.0f;         // races write identical 1.0f
  }
  __syncthreads();
  for (int j = tid; j < NC; j += 128) labels[(size_t)p * NC + j] = lab[j];
  if (tid == 0) maskO[p] = m;
}

// ------------------------------------------------------------- launcher ----
extern "C" void kernel_launch(void* const* d_in, const int* in_sizes, int n_in,
                              void* d_out, int out_size, void* d_ws, size_t ws_size,
                              hipStream_t stream) {
  const float* boxes   = (const float*)d_in[0];
  const float* scores  = (const float*)d_in[1];
  const float* boxes_h = (const float*)d_in[2];
  const float* boxes_o = (const float*)d_in[3];
  const int*   hoi     = (const int*)d_in[4];
  const float* feat    = (const float*)d_in[5];
  const float* W1 = (const float*)d_in[6];
  const float* b1 = (const float*)d_in[7];
  const float* W2 = (const float*)d_in[8];
  const float* b2 = (const float*)d_in[9];
  const float* W3 = (const float*)d_in[10];
  const float* b3 = (const float*)d_in[11];

  // workspace layout (~1.9 MB)
  char* ws = (char*)d_ws;
  float* hm   = (float*)(ws + 0);            // 256 f32
  float* iouh = (float*)(ws + 1024);         // 256x64 f32
  float* iouo = (float*)(ws + 66560);        // 256x64 f32
  unsigned short* W1t = (unsigned short*)(ws + 132096);   // [512][576] bf16
  unsigned short* W2t = (unsigned short*)(ws + 721920);   // [512][512] bf16
  unsigned short* W3t = (unsigned short*)(ws + 1246208);  // [640][512] bf16

  // d_out = logits | labels | pair_mask ; labels region doubles as h1/h2 scratch
  float* logits = (float*)d_out;
  float* labels = logits + (size_t)M_TOT * NC;
  float* maskO  = labels + (size_t)M_TOT * NC;
  unsigned short* h1 = (unsigned short*)labels;           // 67 MB, inside labels region
  unsigned short* h2 = h1 + (size_t)M_TOT * REP;          // next 67 MB (still inside)

  hmask_k<<<1, 256, 0, stream>>>(scores, hm);
  iou_k<<<1, 256, 0, stream>>>(boxes, boxes_h, boxes_o, iouh, iouo);
  convT_k<<<(REP * CF + 255) / 256, 256, 0, stream>>>(W1, W1t, CF, REP, REP);
  convT_k<<<(REP * REP + 255) / 256, 256, 0, stream>>>(W2, W2t, REP, REP, REP);
  convT_k<<<(NC_PAD * REP + 255) / 256, 256, 0, stream>>>(W3, W3t, REP, NC, NC_PAD);

  gemm_k<true, false><<<dim3(REP / BN, M_TOT / BM), 256, 0, stream>>>(
      feat, nullptr, CF, W1t, b1, nullptr, h1, hm);
  gemm_k<false, false><<<dim3(REP / BN, M_TOT / BM), 256, 0, stream>>>(
      nullptr, h1, REP, W2t, b2, nullptr, h2, hm);
  gemm_k<false, true><<<dim3(NC_PAD / BN, M_TOT / BM), 256, 0, stream>>>(
      nullptr, h2, REP, W3t, b3, logits, nullptr, hm);

  labels_k<<<M_TOT, 128, 0, stream>>>(iouh, iouo, hoi, hm, labels, maskO);
}
